// VoxtralRealtimeAttention_83640193122445
// MI455X (gfx1250) — compile-verified, matrix operand order checked by reading, not run on hardware
//
#include <hip/hip_runtime.h>
#include <hip/hip_bf16.h>

// ---------------------------------------------------------------------------
// VoxtralRealtimeAttention for MI455X (gfx1250, wave32, WMMA bf16)
// B=2 S=1500 HID=1280 NH=32 NKV=8 HD=64 WINDOW=750
// ---------------------------------------------------------------------------

typedef __attribute__((ext_vector_type(16))) __bf16 bf16x16;
typedef __attribute__((ext_vector_type(8)))  __bf16 bf16x8;
typedef __attribute__((ext_vector_type(4)))  __bf16 bf16x4;
typedef __attribute__((ext_vector_type(8)))  float  f32x8;

#define Bsz     2
#define Sq      1500
#define SqPad   1504
#define HID     1280
#define NH      32
#define NKV     8
#define HD      64
#define WINDOW  750
#define NEGBIG  (-1.0e9f)

#if defined(__gfx1250__)
#define USE_ASYNC_COPY 1
#else
#define USE_ASYNC_COPY 0
#endif

__device__ __forceinline__ bf16x16 frag_cat(bf16x8 lo, bf16x8 hi) {
  return __builtin_shufflevector(lo, hi, 0,1,2,3,4,5,6,7,8,9,10,11,12,13,14,15);
}

#if USE_ASYNC_COPY
typedef __attribute__((address_space(3))) char lds_char_t;

__device__ __forceinline__ unsigned lds_off(const void* p) {
  // generic -> LDS addrspace cast, then ptrtoint: 32-bit LDS byte address
  return (unsigned)(unsigned long long)(lds_char_t*)p;
}

// ASYNCcnt-tracked DMA: global -> LDS, 16 bytes per lane (VGLOBAL encoding,
// VDST = LDS byte address, VADDR = 64-bit global address)
__device__ __forceinline__ void async_ld_b128(const __bf16* g, __bf16* l) {
  asm volatile("global_load_async_to_lds_b128 %0, %1, off"
               :: "v"(lds_off(l)), "v"(g)
               : "memory");
}

__device__ __forceinline__ void wait_async0() {
  asm volatile("s_wait_asynccnt 0x0" ::: "memory");
}
#endif

// -------------------------- small conversion kernels -----------------------

__global__ __launch_bounds__(256)
void f32_to_bf16_kernel(const float* __restrict__ in, __bf16* __restrict__ out, int n) {
  int i = blockIdx.x * 256 + threadIdx.x;
  if (i < n) out[i] = (__bf16)in[i];
}

// W[K][N] fp32 -> Wt[N][K] bf16
__global__ __launch_bounds__(256)
void transpose_to_bf16_kernel(const float* __restrict__ W, __bf16* __restrict__ Wt,
                              int K, int N) {
  int i = blockIdx.x * 256 + threadIdx.x;
  if (i < K * N) {
    int k = i / N, n = i % N;
    Wt[(size_t)n * K + k] = (__bf16)W[i];
  }
}

// -------------------------- generic bf16 WMMA GEMM -------------------------
// C[M][N] (f32) = A[M][K](bf16) * Bt[N][K](bf16)^T + bias
// block = 256 thr (8 waves), block tile 64(M) x 128(N), K step 32
// wave grid 2(M) x 4(N); each wave computes 32x32 via 4 WMMA accumulators.
__global__ __launch_bounds__(256)
void gemm_bf16_wmma(const __bf16* __restrict__ A, const __bf16* __restrict__ Bt,
                    const float* __restrict__ bias, float* __restrict__ C,
                    int M, int N, int K) {
  __shared__ __align__(16) __bf16 As[64][32];    // 4 KB
  __shared__ __align__(16) __bf16 Bs[128][32];   // 8 KB

  const int t    = threadIdx.x;
  const int lane = t & 31;
  const int wave = t >> 5;
  const int wm   = wave >> 2;     // 0..1
  const int wn   = wave & 3;      // 0..3
  const int m0   = blockIdx.y * 64;
  const int n0   = blockIdx.x * 128;

  // staging coords
  const int ar = t >> 2, ac = (t & 3) * 8;   // A: 8 bf16 (16B) / thread
  const int br = t >> 1, bc = (t & 1) * 16;  // B: 16 bf16 (2x16B) / thread

  // fragment coords (16-bit A/B WMMA layout)
  const int fr = lane & 15;
  const int k0 = (lane < 16) ? 0 : 8;

  f32x8 acc00 = {}, acc01 = {}, acc10 = {}, acc11 = {};

  const bool a_ok = (m0 + ar) < M;
  const __bf16* aGlob = A + (size_t)(m0 + ar) * K + ac;
  const __bf16* bGlob = Bt + (size_t)(n0 + br) * K + bc;

  for (int kt = 0; kt < K; kt += 32) {
#if USE_ASYNC_COPY
    if (a_ok) {
      async_ld_b128(aGlob + kt, &As[ar][ac]);
    } else {
      *(bf16x8*)&As[ar][ac] = (bf16x8){};
    }
    async_ld_b128(bGlob + kt,     &Bs[br][bc]);
    async_ld_b128(bGlob + kt + 8, &Bs[br][bc + 8]);
    if (kt + 32 < K) {
      __builtin_prefetch(aGlob + kt + 32, 0, 1);
      __builtin_prefetch(bGlob + kt + 32, 0, 1);
    }
    wait_async0();
    __syncthreads();
#else
    bf16x8 av = {};
    if (a_ok) av = *(const bf16x8*)(aGlob + kt);
    *(bf16x8*)&As[ar][ac] = av;
    *(bf16x8*)&Bs[br][bc]     = *(const bf16x8*)(bGlob + kt);
    *(bf16x8*)&Bs[br][bc + 8] = *(const bf16x8*)(bGlob + kt + 8);
    __syncthreads();
#endif

    bf16x16 a0 = frag_cat(*(const bf16x8*)&As[wm * 32 + fr][k0],
                          *(const bf16x8*)&As[wm * 32 + fr][k0 + 16]);
    bf16x16 a1 = frag_cat(*(const bf16x8*)&As[wm * 32 + 16 + fr][k0],
                          *(const bf16x8*)&As[wm * 32 + 16 + fr][k0 + 16]);
    bf16x16 b0 = frag_cat(*(const bf16x8*)&Bs[wn * 32 + fr][k0],
                          *(const bf16x8*)&Bs[wn * 32 + fr][k0 + 16]);
    bf16x16 b1 = frag_cat(*(const bf16x8*)&Bs[wn * 32 + 16 + fr][k0],
                          *(const bf16x8*)&Bs[wn * 32 + 16 + fr][k0 + 16]);

    acc00 = __builtin_amdgcn_wmma_f32_16x16x32_bf16(false, a0, false, b0, (short)0, acc00, false, false);
    acc01 = __builtin_amdgcn_wmma_f32_16x16x32_bf16(false, a0, false, b1, (short)0, acc01, false, false);
    acc10 = __builtin_amdgcn_wmma_f32_16x16x32_bf16(false, a1, false, b0, (short)0, acc10, false, false);
    acc11 = __builtin_amdgcn_wmma_f32_16x16x32_bf16(false, a1, false, b1, (short)0, acc11, false, false);
    __syncthreads();
  }

  // store 2x2 tiles of 16x16 (C layout: row=(lane<16?i:8+i), col=lane%16)
  const int colA = n0 + wn * 32 + fr;
  const int colB = colA + 16;
  const float bvA = bias ? bias[colA] : 0.0f;
  const float bvB = bias ? bias[colB] : 0.0f;
#pragma unroll
  for (int i = 0; i < 8; ++i) {
    int r0 = m0 + wm * 32 + ((lane < 16) ? i : 8 + i);
    int r1 = r0 + 16;
    if (r0 < M) {
      C[(size_t)r0 * N + colA] = acc00[i] + bvA;
      C[(size_t)r0 * N + colB] = acc01[i] + bvB;
    }
    if (r1 < M) {
      C[(size_t)r1 * N + colA] = acc10[i] + bvA;
      C[(size_t)r1 * N + colB] = acc11[i] + bvB;
    }
  }
}

// -------------------------- RoPE + pack kernels ----------------------------

// Qf[M][2048] f32 -> RoPE -> Qbf[b][h][s][64] bf16
__global__ __launch_bounds__(256)
void rope_pack_q_kernel(const float* __restrict__ Qf, const float* __restrict__ cosb,
                        const float* __restrict__ sinb, __bf16* __restrict__ Qbf) {
  int i = blockIdx.x * 256 + threadIdx.x;           // over B*S*NH*64 = 6,144,000
  if (i >= Bsz * Sq * NH * HD) return;
  int d  = i & 63;
  int h  = (i >> 6) & 31;
  int bs = i >> 11;                                  // b*S + s
  int b  = bs / Sq, s = bs % Sq;
  const float* row = Qf + (size_t)bs * (NH * HD) + h * HD;
  float c  = cosb[(size_t)bs * HD + d];
  float sn = sinb[(size_t)bs * HD + d];
  float x  = row[d];
  float y  = (d < 32) ? -row[d + 32] : row[d - 32];
  Qbf[(((size_t)(b * NH + h) * Sq) + s) * HD + d] = (__bf16)(x * c + y * sn);
}

// Kf[M][512] f32 -> RoPE -> Kbf[b][kvh][s][64] bf16
__global__ __launch_bounds__(256)
void rope_pack_k_kernel(const float* __restrict__ Kf, const float* __restrict__ cosb,
                        const float* __restrict__ sinb, __bf16* __restrict__ Kbf) {
  int i = blockIdx.x * 256 + threadIdx.x;           // over B*S*NKV*64 = 1,536,000
  if (i >= Bsz * Sq * NKV * HD) return;
  int d  = i & 63;
  int h  = (i >> 6) & 7;
  int bs = i >> 9;
  int b  = bs / Sq, s = bs % Sq;
  const float* row = Kf + (size_t)bs * (NKV * HD) + h * HD;
  float c  = cosb[(size_t)bs * HD + d];
  float sn = sinb[(size_t)bs * HD + d];
  float x  = row[d];
  float y  = (d < 32) ? -row[d + 32] : row[d - 32];
  Kbf[(((size_t)(b * NKV + h) * Sq) + s) * HD + d] = (__bf16)(x * c + y * sn);
}

// Vf[M][512] f32 -> VbfT[b][kvh][64][1504] bf16 (transposed, zero-padded tail)
__global__ __launch_bounds__(256)
void pack_v_kernel(const float* __restrict__ Vf, __bf16* __restrict__ VbfT) {
  int i = blockIdx.x * 256 + threadIdx.x;           // over B*NKV*64*1504 = 1,540,096
  if (i >= Bsz * NKV * HD * SqPad) return;
  int sp   = i % SqPad;
  int rest = i / SqPad;
  int d = rest & 63;
  int h = (rest >> 6) & 7;
  int b = rest >> 9;
  float v = 0.0f;
  if (sp < Sq) v = Vf[(size_t)(b * Sq + sp) * (NKV * HD) + h * HD + d];
  VbfT[((size_t)(b * NKV + h) * HD + d) * SqPad + sp] = (__bf16)v;
}

// -------------------------- flash attention (WMMA) -------------------------
// one wave32 per (16-query tile, head, batch)
__global__ __launch_bounds__(32)
void attn_fwd_kernel(const __bf16* __restrict__ Qbf, const __bf16* __restrict__ Kbf,
                     const __bf16* __restrict__ VbfT, __bf16* __restrict__ Obf) {
  __shared__ __align__(16) __bf16 P[16][32];

  const int lane = threadIdx.x;
  const int qt = blockIdx.x, h = blockIdx.y, b = blockIdx.z;
  const int kvh = h >> 2;                 // NH/NKV = 4
  const int q0  = qt * 16;
  const int fr  = lane & 15;
  const int k0  = (lane < 16) ? 0 : 8;
  const float scale = 0.125f;             // HD^-0.5

  // ---- Q fragments: A-layout, 2 chunks over HD=64 (K=32 each)
  int qrow = q0 + fr; if (qrow > Sq - 1) qrow = Sq - 1;
  const __bf16* qp = Qbf + ((size_t)(b * NH + h) * Sq + qrow) * HD;
  bf16x16 aq0 = frag_cat(*(const bf16x8*)(qp + k0),      *(const bf16x8*)(qp + k0 + 16));
  bf16x16 aq1 = frag_cat(*(const bf16x8*)(qp + 32 + k0), *(const bf16x8*)(qp + 32 + k0 + 16));

  float m_i[8], l_i[8];
#pragma unroll
  for (int i = 0; i < 8; ++i) { m_i[i] = -3.0e38f; l_i[i] = 0.0f; }
  f32x8 o[4] = {};

  int kbeg = q0 - (WINDOW - 1); if (kbeg < 0) kbeg = 0;
  kbeg &= ~31;
  int kend = q0 + 15; if (kend > Sq - 1) kend = Sq - 1;

  const __bf16* kbase = Kbf + (size_t)(b * NKV + kvh) * Sq * HD;
  const __bf16* vbase = VbfT + (size_t)(b * NKV + kvh) * HD * SqPad;

  for (int kt = kbeg; kt <= kend; kt += 32) {
    f32x8 s0 = {}, s1 = {};
    {
      int key = kt + fr; if (key > Sq - 1) key = Sq - 1;
      const __bf16* kp = kbase + (size_t)key * HD;
      bf16x16 b0 = frag_cat(*(const bf16x8*)(kp + k0),      *(const bf16x8*)(kp + k0 + 16));
      bf16x16 b1 = frag_cat(*(const bf16x8*)(kp + 32 + k0), *(const bf16x8*)(kp + 32 + k0 + 16));
      s0 = __builtin_amdgcn_wmma_f32_16x16x32_bf16(false, aq0, false, b0, (short)0, s0, false, false);
      s0 = __builtin_amdgcn_wmma_f32_16x16x32_bf16(false, aq1, false, b1, (short)0, s0, false, false);
    }
    {
      int key = kt + 16 + fr; if (key > Sq - 1) key = Sq - 1;
      const __bf16* kp = kbase + (size_t)key * HD;
      bf16x16 b0 = frag_cat(*(const bf16x8*)(kp + k0),      *(const bf16x8*)(kp + k0 + 16));
      bf16x16 b1 = frag_cat(*(const bf16x8*)(kp + 32 + k0), *(const bf16x8*)(kp + 32 + k0 + 16));
      s1 = __builtin_amdgcn_wmma_f32_16x16x32_bf16(false, aq0, false, b0, (short)0, s1, false, false);
      s1 = __builtin_amdgcn_wmma_f32_16x16x32_bf16(false, aq1, false, b1, (short)0, s1, false, false);
    }

    // ---- scale + sliding-window causal mask (C layout: row=(lane<16?i:8+i), col=lane%16)
    const int qb = q0 + ((lane < 16) ? 0 : 8);
    const int kk0c = kt + fr;
    const int kk1c = kt + 16 + fr;
#pragma unroll
    for (int i = 0; i < 8; ++i) {
      int qq = qb + i;
      float v0 = s0[i] * scale;
      float v1 = s1[i] * scale;
      bool ok0 = (qq >= kk0c) && (qq - kk0c < WINDOW) && (qq < Sq) && (kk0c < Sq);
      bool ok1 = (qq >= kk1c) && (qq - kk1c < WINDOW) && (qq < Sq) && (kk1c < Sq);
      s0[i] = ok0 ? v0 : (v0 + NEGBIG);
      s1[i] = ok1 ? v1 : (v1 + NEGBIG);
    }

    // ---- online softmax over the 32-key tile
#pragma unroll
    for (int i = 0; i < 8; ++i) {
      float mx = fmaxf(s0[i], s1[i]);
#pragma unroll
      for (int off = 1; off < 16; off <<= 1) mx = fmaxf(mx, __shfl_xor(mx, off, 32));
      float nm = fmaxf(m_i[i], mx);
      float a  = __expf(m_i[i] - nm);
      float e0 = __expf(s0[i] - nm);
      float e1 = __expf(s1[i] - nm);
      float sum = e0 + e1;
#pragma unroll
      for (int off = 1; off < 16; off <<= 1) sum += __shfl_xor(sum, off, 32);
      l_i[i] = l_i[i] * a + sum;
      m_i[i] = nm;
      o[0][i] *= a; o[1][i] *= a; o[2][i] *= a; o[3][i] *= a;
      s0[i] = e0; s1[i] = e1;
    }

    // ---- P: C-layout f32 -> LDS row-major bf16 -> A-layout fragment
    __syncthreads();
#pragma unroll
    for (int i = 0; i < 8; ++i) {
      int row = (lane < 16) ? i : 8 + i;
      P[row][fr]      = (__bf16)s0[i];
      P[row][16 + fr] = (__bf16)s1[i];
    }
    __syncthreads();
    bf16x16 ap = frag_cat(*(const bf16x8*)&P[fr][k0], *(const bf16x8*)&P[fr][k0 + 16]);

    // ---- PV: 4 HD chunks of 16, contraction over 32 keys
#pragma unroll
    for (int c = 0; c < 4; ++c) {
      const __bf16* vp = vbase + (size_t)(c * 16 + fr) * SqPad + kt;
      bf16x16 bv = frag_cat(*(const bf16x8*)(vp + k0), *(const bf16x8*)(vp + k0 + 16));
      o[c] = __builtin_amdgcn_wmma_f32_16x16x32_bf16(false, ap, false, bv, (short)0, o[c], false, false);
    }
  }

  // ---- finalize: O/l -> attn bf16 [b][s][h*64+d]
#pragma unroll
  for (int i = 0; i < 8; ++i) {
    int qq = q0 + ((lane < 16) ? i : 8 + i);
    if (qq < Sq) {
      float inv = 1.0f / l_i[i];
      size_t base = ((size_t)(b * Sq + qq)) * (NH * HD) + h * HD + fr;
      Obf[base +  0] = (__bf16)(o[0][i] * inv);
      Obf[base + 16] = (__bf16)(o[1][i] * inv);
      Obf[base + 32] = (__bf16)(o[2][i] * inv);
      Obf[base + 48] = (__bf16)(o[3][i] * inv);
    }
  }
}

// -------------------------- workspace layout (bytes) -----------------------
#define OFF_XBF  ((size_t)0)            // 3000*1280*2  =  7,680,000
#define OFF_WQT  ((size_t)7680000)      // 2048*1280*2  =  5,242,880
#define OFF_WKT  ((size_t)12922880)     //  512*1280*2  =  1,310,720
#define OFF_WVT  ((size_t)14233600)     //  512*1280*2  =  1,310,720
#define OFF_WOT  ((size_t)15544320)     // 1280*2048*2  =  5,242,880
#define OFF_QF   ((size_t)20787200)     // 3000*2048*4  = 24,576,000
#define OFF_KF   ((size_t)45363200)     // 3000*512*4   =  6,144,000
#define OFF_VF   ((size_t)51507200)     // 3000*512*4   =  6,144,000
#define OFF_QBF  ((size_t)57651200)     // 2*32*1500*64*2  = 12,288,000
#define OFF_KBF  ((size_t)69939200)     // 2*8*1500*64*2   =  3,072,000
#define OFF_VBT  ((size_t)73011200)     // 2*8*64*1504*2   =  3,080,192
#define OFF_ABF  ((size_t)76091392)     // 3000*2048*2     = 12,288,000
// total = 88,379,392 bytes

extern "C" void kernel_launch(void* const* d_in, const int* in_sizes, int n_in,
                              void* d_out, int out_size, void* d_ws, size_t ws_size,
                              hipStream_t stream) {
  const float* hs   = (const float*)d_in[0];
  const float* cosb = (const float*)d_in[1];
  const float* sinb = (const float*)d_in[2];
  // d_in[3] = attention_mask, computed analytically instead
  const float* Wq = (const float*)d_in[4];
  const float* bq = (const float*)d_in[5];
  const float* Wk = (const float*)d_in[6];
  const float* Wv = (const float*)d_in[7];
  const float* bv = (const float*)d_in[8];
  const float* Wo = (const float*)d_in[9];
  const float* bo = (const float*)d_in[10];

  char* ws = (char*)d_ws;
  __bf16* Xbf  = (__bf16*)(ws + OFF_XBF);
  __bf16* WqT  = (__bf16*)(ws + OFF_WQT);
  __bf16* WkT  = (__bf16*)(ws + OFF_WKT);
  __bf16* WvT  = (__bf16*)(ws + OFF_WVT);
  __bf16* WoT  = (__bf16*)(ws + OFF_WOT);
  float*  Qf   = (float*)(ws + OFF_QF);
  float*  Kf   = (float*)(ws + OFF_KF);
  float*  Vf   = (float*)(ws + OFF_VF);
  __bf16* Qbf  = (__bf16*)(ws + OFF_QBF);
  __bf16* Kbf  = (__bf16*)(ws + OFF_KBF);
  __bf16* VbfT = (__bf16*)(ws + OFF_VBT);
  __bf16* Abf  = (__bf16*)(ws + OFF_ABF);

  const int M = Bsz * Sq;                 // 3000

  // 1) convert inputs to bf16 (weights transposed to [N][K])
  {
    int n = M * HID;                      // 3,840,000
    f32_to_bf16_kernel<<<(n + 255) / 256, 256, 0, stream>>>(hs, Xbf, n);
    transpose_to_bf16_kernel<<<(HID * NH * HD + 255) / 256, 256, 0, stream>>>(Wq, WqT, HID, NH * HD);
    transpose_to_bf16_kernel<<<(HID * NKV * HD + 255) / 256, 256, 0, stream>>>(Wk, WkT, HID, NKV * HD);
    transpose_to_bf16_kernel<<<(HID * NKV * HD + 255) / 256, 256, 0, stream>>>(Wv, WvT, HID, NKV * HD);
    transpose_to_bf16_kernel<<<(NH * HD * HID + 255) / 256, 256, 0, stream>>>(Wo, WoT, NH * HD, HID);
  }

  // 2) QKV projections (WMMA GEMMs)
  {
    dim3 blk(256);
    dim3 gq((NH * HD) / 128, (M + 63) / 64);     // 16 x 47
    gemm_bf16_wmma<<<gq, blk, 0, stream>>>(Xbf, WqT, bq, Qf, M, NH * HD, HID);
    dim3 gk((NKV * HD) / 128, (M + 63) / 64);    // 4 x 47
    gemm_bf16_wmma<<<gk, blk, 0, stream>>>(Xbf, WkT, nullptr, Kf, M, NKV * HD, HID);
    gemm_bf16_wmma<<<gk, blk, 0, stream>>>(Xbf, WvT, bv, Vf, M, NKV * HD, HID);
  }

  // 3) RoPE + repack to attention layouts
  {
    int nq = Bsz * Sq * NH * HD;          // 6,144,000
    rope_pack_q_kernel<<<(nq + 255) / 256, 256, 0, stream>>>(Qf, cosb, sinb, Qbf);
    int nk = Bsz * Sq * NKV * HD;         // 1,536,000
    rope_pack_k_kernel<<<(nk + 255) / 256, 256, 0, stream>>>(Kf, cosb, sinb, Kbf);
    int nv = Bsz * NKV * HD * SqPad;      // 1,540,096
    pack_v_kernel<<<(nv + 255) / 256, 256, 0, stream>>>(Vf, VbfT);
  }

  // 4) sliding-window flash attention
  {
    dim3 grid((Sq + 15) / 16, NH, Bsz);   // 94 x 32 x 2
    attn_fwd_kernel<<<grid, 32, 0, stream>>>(Qbf, Kbf, VbfT, Abf);
  }

  // 5) output projection -> d_out (f32)
  {
    dim3 go(HID / 128, (M + 63) / 64);    // 10 x 47
    gemm_bf16_wmma<<<go, 256, 0, stream>>>(Abf, WoT, bo, (float*)d_out, M, HID, NH * HD);
  }
}
